// Model_17188459119284
// MI455X (gfx1250) — compile-verified, matrix operand order checked by reading, not run on hardware
//
#include <hip/hip_runtime.h>
#include <hip/hip_bf16.h>

// ---------------- types & helpers ----------------
typedef __attribute__((ext_vector_type(8)))  float  v8f;
typedef __attribute__((ext_vector_type(16))) __bf16 v16bf;
typedef __attribute__((ext_vector_type(8)))  __bf16 v8bf;
typedef int v4i __attribute__((vector_size(16)));
typedef __attribute__((address_space(1))) v4i as1_v4i;
typedef __attribute__((address_space(3))) v4i as3_v4i;

#define TD    1024      // model dim D
#define TS    1024      // seq len S
#define TB    8         // batch
#define TH    16        // heads
#define TDH   64        // head dim
#define TROWS (TB*TS)   // 8192 token rows
#define TFF   4096      // ffn dim
#define NOUT  32000
#define NLAY  4
#define SCW   1032      // score LDS row stride (1024 + 8 pad)

static __device__ __forceinline__ unsigned short f2bf(float f) {
  union { float f; unsigned u; } x; x.f = f;
  unsigned r = x.u + 0x7FFFu + ((x.u >> 16) & 1u);
  return (unsigned short)(r >> 16);
}
static __device__ __forceinline__ __bf16 f2bf16(float f) {
  union { unsigned short s; __bf16 b; } y; y.s = f2bf(f);
  return y.b;
}
static __device__ __forceinline__ v8f zero8() {
  v8f z;
#pragma unroll
  for (int i = 0; i < 8; i++) z[i] = 0.0f;
  return z;
}
// Async global->LDS 16B copy (CDNA5 GLOBAL_LOAD_ASYNC_TO_LDS_B128, ASYNCcnt).
static __device__ __forceinline__ void async_copy16(const void* g, void* l) {
#if __has_builtin(__builtin_amdgcn_global_load_async_to_lds_b128)
  __builtin_amdgcn_global_load_async_to_lds_b128(
      (as1_v4i*)g, (as3_v4i*)l, 0, 0);
#else
  unsigned lo = (unsigned)(unsigned long long)l;
  asm volatile("global_load_async_to_lds_b128 %0, %1, off"
               :: "v"(lo), "v"(g) : "memory");
#endif
}
static __device__ __forceinline__ void wait_async0() {
#if __has_builtin(__builtin_amdgcn_s_wait_asynccnt)
  __builtin_amdgcn_s_wait_asynccnt(0);
#else
  asm volatile("s_wait_asynccnt 0" ::: "memory");
#endif
}
// A/B fragment loader. base = tile origin (row-major, rows of `ld` ushorts).
// A: lane = M row; B: lane = N row ([N,K] memory). Chunks per ISA 7.12.2:
// lanes 0-15 -> K {0..7, 16..23}; lanes 16-31 -> K {8..15, 24..31}.
static __device__ __forceinline__ v16bf load_frag(const unsigned short* base,
                                                  size_t ld, int lane) {
  const int hl = lane >> 4, lr = lane & 15;
  const unsigned short* p = base + (size_t)lr * ld + hl * 8;
  union { v16bf v; v8bf h[2]; } f;
  f.h[0] = *reinterpret_cast<const v8bf*>(p);
  f.h[1] = *reinterpret_cast<const v8bf*>(p + 16);
  return f.v;
}
static __device__ __forceinline__ v8f wmma_bf(v16bf a, v16bf b, v8f c) {
  return __builtin_amdgcn_wmma_f32_16x16x32_bf16(false, a, false, b,
                                                 (short)0, c, false, false);
}

// ---------------- fp32 -> bf16 weight conversion ----------------
__global__ __launch_bounds__(256) void cvt_bf16_kernel(
    const float* __restrict__ src, unsigned short* __restrict__ dst, long n) {
  long i = ((long)blockIdx.x * 256 + threadIdx.x) * 4;
  if (i + 4 <= n) {
    float4 f = *reinterpret_cast<const float4*>(src + i);
    unsigned short r[4] = {f2bf(f.x), f2bf(f.y), f2bf(f.z), f2bf(f.w)};
    *reinterpret_cast<uint2*>(dst + i) = *reinterpret_cast<const uint2*>(r);
  }
}

// ---------------- embedding gather ----------------
__global__ __launch_bounds__(256) void embed_kernel(
    const float* __restrict__ EW, const int* __restrict__ x,
    float* __restrict__ h32, unsigned short* __restrict__ h16) {
  int row = blockIdx.x;
  int tok = x[row];
  const float* src = EW + (size_t)tok * TD;
  size_t base = (size_t)row * TD;
  for (int c = threadIdx.x; c < TD; c += 256) {
    float v = src[c];
    h32[base + c] = v;
    h16[base + c] = f2bf(v);
  }
}

// ---------------- generic bf16 GEMM: C = A[M,K] * B[N,K]^T ----------------
// Block tile 128x256, K-step 32. Double-buffered LDS fed by async
// global->LDS DMA (no staging VGPRs, ASYNCcnt-tracked), 1 barrier/step.
// 8 waves in a 2x4 grid; each wave owns a 64x64 tile = 4x4 WMMA accumulators
// -> 16 v_wmma per K-step against 8 fragment loads (4x register reuse).
__global__ __launch_bounds__(256, 1) void gemm_bf16_kernel(
    const unsigned short* __restrict__ A,   // [M,K] bf16
    const unsigned short* __restrict__ Bw,  // [N,K] bf16
    const float* __restrict__ bias,         // [N] or null
    float* __restrict__ C32,                // [M,N] or null
    unsigned short* __restrict__ C16,       // [M,N] or null
    unsigned short* __restrict__ C16T,      // [N,M] or null (transposed out)
    int M, int N, int K, int relu) {
  __shared__ unsigned short As[2][128 * 40];
  __shared__ unsigned short Bs[2][256 * 40];
  const int tid = threadIdx.x, w = tid >> 5, lane = tid & 31;
  const int bm = blockIdx.y * 128, bn = blockIdx.x * 256;
  const int wm = (w & 1) * 64, wn = (w >> 1) * 64;
  const int sr = tid >> 2, sc = (tid & 3) * 8;   // staging row/col

  v8f acc[4][4];
#pragma unroll
  for (int i = 0; i < 4; i++)
#pragma unroll
    for (int j = 0; j < 4; j++) acc[i][j] = zero8();

  auto issue = [&](int buf, int k0) {   // async DMA of next 128x32/256x32 tile
#pragma unroll
    for (int i = 0; i < 2; i++)
      async_copy16(&A[(size_t)(bm + sr + i * 64) * K + k0 + sc],
                   &As[buf][(sr + i * 64) * 40 + sc]);
#pragma unroll
    for (int j = 0; j < 4; j++)
      async_copy16(&Bw[(size_t)(bn + sr + j * 64) * K + k0 + sc],
                   &Bs[buf][(sr + j * 64) * 40 + sc]);
  };

  issue(0, 0);
  wait_async0();
  __syncthreads();

  const int nsteps = K >> 5;
  for (int s = 0; s < nsteps; s++) {
    const int buf = s & 1;
    if (s + 1 < nsteps) issue(buf ^ 1, (s + 1) << 5);  // DMA next tile
    v16bf af[4], bfr[4];
#pragma unroll
    for (int mi = 0; mi < 4; mi++)
      af[mi] = load_frag(&As[buf][(wm + mi * 16) * 40], 40, lane);
#pragma unroll
    for (int ni = 0; ni < 4; ni++)
      bfr[ni] = load_frag(&Bs[buf][(wn + ni * 16) * 40], 40, lane);
#pragma unroll
    for (int mi = 0; mi < 4; mi++)
#pragma unroll
      for (int ni = 0; ni < 4; ni++)
        acc[mi][ni] = wmma_bf(af[mi], bfr[ni], acc[mi][ni]);
    if (s + 1 < nsteps) wait_async0();   // own DMA done before publishing
    __syncthreads();                     // single barrier per K-step
  }

  const int hl = lane >> 4, lr = lane & 15;
#pragma unroll
  for (int mi = 0; mi < 4; mi++)
#pragma unroll
    for (int ni = 0; ni < 4; ni++) {
      int col = bn + wn + ni * 16 + lr;
      float bv = bias ? bias[col] : 0.0f;
#pragma unroll
      for (int v = 0; v < 8; v++) {
        int row = bm + wm + mi * 16 + v + hl * 8;
        float o = acc[mi][ni][v] + bv;
        if (relu) o = fmaxf(o, 0.0f);
        if (C32)  C32 [(size_t)row * N + col] = o;
        if (C16)  C16 [(size_t)row * N + col] = f2bf(o);
        if (C16T) C16T[(size_t)col * M + row] = f2bf(o);
      }
    }
}

// ---------------- fused causal attention ----------------
// One workgroup per (q-block of 64 rows, head, batch). Full 64x1024 fp32
// score strip in LDS (dynamic, ~258KB of the WGP's 320KB).
__global__ __launch_bounds__(256) void attn_kernel(
    const unsigned short* __restrict__ Q,   // [B*S, D] bf16
    const unsigned short* __restrict__ Kb,  // [B*S, D] bf16
    const unsigned short* __restrict__ Vt,  // [D, B*S] bf16 (transposed V)
    float* __restrict__ O) {                // [B*S, D] f32
  extern __shared__ float sc[];             // [64][SCW]
  const int qb = blockIdx.x, h = blockIdx.y, b = blockIdx.z;
  const int q0 = qb * 64;
  const int tid = threadIdx.x, w = tid >> 5, lane = tid & 31;
  const int ncols = q0 + 64;                // causal horizon for this block
  const int hl = lane >> 4, lr = lane & 15;
  const size_t rowbase = (size_t)b * TS;

  // ---- phase 1: scores = (Q K^T) * 1/sqrt(DH) + causal mask -> LDS ----
  {
    const int qt = w >> 1, halfn = w & 1;
    const unsigned short* qtile =
        Q + (rowbase + q0 + qt * 16) * TD + h * TDH;
    v16bf A0 = load_frag(qtile, TD, lane);        // dh 0..31
    v16bf A1 = load_frag(qtile + 32, TD, lane);   // dh 32..63
    const int ntiles = ncols >> 4;
    for (int nt = halfn; nt < ntiles; nt += 2) {
      const int n0 = nt * 16;
      const unsigned short* ktile =
          Kb + (rowbase + n0) * TD + h * TDH;
      v16bf B0 = load_frag(ktile, TD, lane);
      v16bf B1 = load_frag(ktile + 32, TD, lane);
      v8f acc = zero8();
      acc = wmma_bf(A0, B0, acc);
      acc = wmma_bf(A1, B1, acc);
      const int ncol = n0 + lr;
#pragma unroll
      for (int v = 0; v < 8; v++) {
        const int mrow = qt * 16 + v + hl * 8;
        float s = acc[v] * 0.125f;                // 1/sqrt(64)
        if (ncol > q0 + mrow) s += -1.25e8f;      // (-1e9)/sqrt(64)
        sc[mrow * SCW + ncol] = s;
      }
    }
  }
  __syncthreads();

  // ---- phase 2: row softmax over [0, ncols) in LDS ----
  for (int r = w * 8; r < w * 8 + 8; r++) {
    float mx = -3.4e38f;
    for (int c = lane; c < ncols; c += 32) mx = fmaxf(mx, sc[r * SCW + c]);
#pragma unroll
    for (int off = 16; off > 0; off >>= 1) mx = fmaxf(mx, __shfl_xor(mx, off, 32));
    float sum = 0.0f;
    for (int c = lane; c < ncols; c += 32) {
      float e = __expf(sc[r * SCW + c] - mx);
      sc[r * SCW + c] = e;
      sum += e;
    }
#pragma unroll
    for (int off = 16; off > 0; off >>= 1) sum += __shfl_xor(sum, off, 32);
    float inv = 1.0f / sum;
    for (int c = lane; c < ncols; c += 32) sc[r * SCW + c] *= inv;
  }
  __syncthreads();

  // ---- phase 3: O = P @ V (A from LDS fp32->bf16, B from Vt) ----
  {
    const int qt = w & 3, nh = w >> 2;     // 16 q-rows, 32 out cols per wave
    v8f acc0 = zero8(), acc1 = zero8();
    for (int kk = 0; kk < ncols; kk += 32) {
      v16bf av;
      const float* prow = &sc[(qt * 16 + lr) * SCW + kk];
#pragma unroll
      for (int j = 0; j < 8; j++) {
        av[j]     = f2bf16(prow[hl * 8 + j]);
        av[8 + j] = f2bf16(prow[16 + hl * 8 + j]);
      }
      const unsigned short* vt0 =
          Vt + (size_t)(h * TDH + nh * 32) * TROWS + rowbase + kk;
      v16bf B0 = load_frag(vt0, TROWS, lane);
      v16bf B1 = load_frag(vt0 + (size_t)16 * TROWS, TROWS, lane);
      acc0 = wmma_bf(av, B0, acc0);
      acc1 = wmma_bf(av, B1, acc1);
    }
#pragma unroll
    for (int v = 0; v < 8; v++) {
      const size_t row = rowbase + q0 + qt * 16 + v + hl * 8;
      O[row * TD + h * TDH + nh * 32 + lr]      = acc0[v];
      O[row * TD + h * TDH + nh * 32 + 16 + lr] = acc1[v];
    }
  }
}

// ---------------- residual add + layernorm (fp32 + bf16 out) -------------
__global__ __launch_bounds__(256) void add_ln_kernel(
    const float* __restrict__ a, const float* __restrict__ r,
    const float* __restrict__ g, const float* __restrict__ be,
    float* __restrict__ y32, unsigned short* __restrict__ y16) {
  __shared__ float red[256];
  const int tid = threadIdx.x;
  const size_t base = (size_t)blockIdx.x * TD;
  float x[4];
  float s = 0.0f;
#pragma unroll
  for (int j = 0; j < 4; j++) {
    int c = tid + j * 256;
    x[j] = a[base + c] + r[base + c];
    s += x[j];
  }
  red[tid] = s; __syncthreads();
  for (int st = 128; st > 0; st >>= 1) {
    if (tid < st) red[tid] += red[tid + st];
    __syncthreads();
  }
  const float mean = red[0] * (1.0f / TD);
  __syncthreads();
  float s2 = 0.0f;
#pragma unroll
  for (int j = 0; j < 4; j++) { float d = x[j] - mean; s2 += d * d; }
  red[tid] = s2; __syncthreads();
  for (int st = 128; st > 0; st >>= 1) {
    if (tid < st) red[tid] += red[tid + st];
    __syncthreads();
  }
  const float inv = rsqrtf(red[0] * (1.0f / TD) + 1e-5f);
#pragma unroll
  for (int j = 0; j < 4; j++) {
    int c = tid + j * 256;
    float yv = (x[j] - mean) * inv * g[c] + be[c];
    y32[base + c] = yv;
    y16[base + c] = f2bf(yv);
  }
}

// ---------------- final logits: [8,1024] @ [32000,1024]^T + b ------------
__global__ __launch_bounds__(256) void logits_kernel(
    const float* __restrict__ h32, const float* __restrict__ Wf,
    const float* __restrict__ bf, float* __restrict__ logits) {
  const int n = blockIdx.x * 8 + (threadIdx.x >> 5);
  const int lane = threadIdx.x & 31;
  const float* wrow = Wf + (size_t)n * TD;
  float acc[TB];
#pragma unroll
  for (int b = 0; b < TB; b++) acc[b] = 0.0f;
  for (int c = lane; c < TD; c += 32) {
    if (c + 256 < TD) __builtin_prefetch(wrow + c + 256, 0, 0);
    float wv = wrow[c];
#pragma unroll
    for (int b = 0; b < TB; b++)
      acc[b] += wv * h32[((size_t)b * TS + (TS - 1)) * TD + c];
  }
#pragma unroll
  for (int b = 0; b < TB; b++) {
#pragma unroll
    for (int off = 16; off > 0; off >>= 1) acc[b] += __shfl_xor(acc[b], off, 32);
    if (lane == 0) logits[(size_t)b * NOUT + n] = acc[b] + bf[n];
  }
}

// ---------------- per-batch log-softmax term + final loss ---------------
__global__ __launch_bounds__(256) void lse_kernel(
    const float* __restrict__ logits, const int* __restrict__ label,
    float* __restrict__ terms) {
  __shared__ float red[256];
  const int tid = threadIdx.x, b = blockIdx.x;
  const float* row = logits + (size_t)b * NOUT;
  float mx = -3.4e38f;
  for (int c = tid; c < NOUT; c += 256) mx = fmaxf(mx, row[c]);
  red[tid] = mx; __syncthreads();
  for (int st = 128; st > 0; st >>= 1) {
    if (tid < st) red[tid] = fmaxf(red[tid], red[tid + st]);
    __syncthreads();
  }
  mx = red[0]; __syncthreads();
  float sum = 0.0f;
  for (int c = tid; c < NOUT; c += 256) sum += __expf(row[c] - mx);
  red[tid] = sum; __syncthreads();
  for (int st = 128; st > 0; st >>= 1) {
    if (tid < st) red[tid] += red[tid + st];
    __syncthreads();
  }
  if (tid == 0) terms[b] = row[label[b]] - (mx + __logf(red[0]));
}

__global__ void loss_kernel(const float* __restrict__ terms,
                            float* __restrict__ out) {
  if (threadIdx.x == 0) {
    float s = 0.0f;
    for (int b = 0; b < TB; b++) s += terms[b];
    out[0] = -s * (1.0f / TB);
  }
}

// ---------------- host orchestration ----------------
extern "C" void kernel_launch(void* const* d_in, const int* in_sizes, int n_in,
                              void* d_out, int out_size, void* d_ws,
                              size_t ws_size, hipStream_t stream) {
  (void)in_sizes; (void)n_in; (void)out_size; (void)ws_size;
  const float* embed_W = (const float*)d_in[0];
  const float* Wq      = (const float*)d_in[1];
  const float* Wk      = (const float*)d_in[2];
  const float* Wv      = (const float*)d_in[3];
  const float* w1_W    = (const float*)d_in[4];
  const float* w1_b    = (const float*)d_in[5];
  const float* w2_W    = (const float*)d_in[6];
  const float* w2_b    = (const float*)d_in[7];
  const float* ln1_g   = (const float*)d_in[8];
  const float* ln1_b   = (const float*)d_in[9];
  const float* ln2_g   = (const float*)d_in[10];
  const float* ln2_b   = (const float*)d_in[11];
  const float* final_W = (const float*)d_in[12];
  const float* final_b = (const float*)d_in[13];
  const int*   x       = (const int*)d_in[14];
  const int*   label   = (const int*)d_in[15];
  // d_in[16] = src_mask: folded analytically into attn_kernel.

  char* ws = (char*)d_ws;
  size_t off = 0;
  auto alloc = [&](size_t bytes) -> void* {
    void* p = ws + off;
    off += (bytes + 255) & ~(size_t)255;
    return p;
  };
  const size_t DD = (size_t)TD * TD;
  unsigned short* WqB  = (unsigned short*)alloc(NLAY * DD * 2);
  unsigned short* WkB  = (unsigned short*)alloc(NLAY * DD * 2);
  unsigned short* WvB  = (unsigned short*)alloc(NLAY * DD * 2);
  unsigned short* W1B  = (unsigned short*)alloc((size_t)NLAY * TFF * TD * 2);
  unsigned short* W2B  = (unsigned short*)alloc((size_t)NLAY * TD * TFF * 2);
  float*          h32  = (float*)alloc((size_t)TROWS * TD * 4);
  unsigned short* h16  = (unsigned short*)alloc((size_t)TROWS * TD * 2);
  unsigned short* q16  = (unsigned short*)alloc((size_t)TROWS * TD * 2);
  unsigned short* k16  = (unsigned short*)alloc((size_t)TROWS * TD * 2);
  unsigned short* vT   = (unsigned short*)alloc((size_t)TD * TROWS * 2);
  float*          o32  = (float*)alloc((size_t)TROWS * TD * 4);
  float*          y132 = (float*)alloc((size_t)TROWS * TD * 4);
  unsigned short* y116 = (unsigned short*)alloc((size_t)TROWS * TD * 2);
  unsigned short* f116 = (unsigned short*)alloc((size_t)TROWS * TFF * 2);
  float*          f232 = (float*)alloc((size_t)TROWS * TD * 4);
  float*          lg   = (float*)alloc((size_t)TB * NOUT * 4);
  float*          terms= (float*)alloc(TB * 4);

  // weights -> bf16
  auto cvt = [&](const float* s, unsigned short* d, long n) {
    cvt_bf16_kernel<<<dim3((unsigned)(n / 1024)), 256, 0, stream>>>(s, d, n);
  };
  cvt(Wq,   WqB, (long)NLAY * DD);
  cvt(Wk,   WkB, (long)NLAY * DD);
  cvt(Wv,   WvB, (long)NLAY * DD);
  cvt(w1_W, W1B, (long)NLAY * TFF * TD);
  cvt(w2_W, W2B, (long)NLAY * TD * TFF);

  embed_kernel<<<TROWS, 256, 0, stream>>>(embed_W, x, h32, h16);

  const dim3 gD (TD  / 256, TROWS / 128);   // N=1024 GEMMs: 4 x 64 blocks
  const dim3 gFF(TFF / 256, TROWS / 128);   // N=4096 GEMM: 16 x 64 blocks
  const unsigned scBytes = 64 * SCW * 4;    // 264192 B dynamic LDS

  for (int l = 0; l < NLAY; l++) {
    // QKV projections (bf16 out; V also stored transposed for attention)
    gemm_bf16_kernel<<<gD, 256, 0, stream>>>(
        h16, WqB + (size_t)l * DD, nullptr, nullptr, q16, nullptr,
        TROWS, TD, TD, 0);
    gemm_bf16_kernel<<<gD, 256, 0, stream>>>(
        h16, WkB + (size_t)l * DD, nullptr, nullptr, k16, nullptr,
        TROWS, TD, TD, 0);
    gemm_bf16_kernel<<<gD, 256, 0, stream>>>(
        h16, WvB + (size_t)l * DD, nullptr, nullptr, nullptr, vT,
        TROWS, TD, TD, 0);
    // fused causal attention
    attn_kernel<<<dim3(TS / 64, TH, TB), 256, scBytes, stream>>>(
        q16, k16, vT, o32);
    // LN1(o + h)
    add_ln_kernel<<<TROWS, 256, 0, stream>>>(
        o32, h32, ln1_g + l * TD, ln1_b + l * TD, y132, y116);
    // FFN
    gemm_bf16_kernel<<<gFF, 256, 0, stream>>>(
        y116, W1B + (size_t)l * TFF * TD, w1_b + (size_t)l * TFF,
        nullptr, f116, nullptr, TROWS, TFF, TD, 1);
    gemm_bf16_kernel<<<gD, 256, 0, stream>>>(
        f116, W2B + (size_t)l * TD * TFF, w2_b + (size_t)l * TD,
        f232, nullptr, nullptr, TROWS, TD, TFF, 0);
    // LN2(f + y1) -> new residual stream
    add_ln_kernel<<<TROWS, 256, 0, stream>>>(
        f232, y132, ln2_g + l * TD, ln2_b + l * TD, h32, h16);
  }

  logits_kernel<<<NOUT / 8, 256, 0, stream>>>(h32, final_W, final_b, lg);
  lse_kernel<<<TB, 256, 0, stream>>>(lg, label, terms);
  loss_kernel<<<1, 32, 0, stream>>>(terms, (float*)d_out);
}
